// MixtureOfExpertsLayer_41669772706425
// MI455X (gfx1250) — compile-verified, matrix operand order checked by reading, not run on hardware
//
#include <hip/hip_runtime.h>
#include <hip/hip_bf16.h>
#include <cstdint>

// Problem constants (match reference)
#define B_TOK   8192
#define D_DIM   1024
#define E_EXP   8
#define H_DIM   4096
#define TOP_K   2
#define TM      32          // token tile (M) per block
#define PENALTY 0.01f

typedef __attribute__((ext_vector_type(16))) __bf16 v16bf;
typedef __attribute__((ext_vector_type(8)))  float  v8f;

union Frag16 {              // one wmma bf16 operand fragment (per lane)
    v16bf  v;
    __bf16 x[16];
    uint4  q[2];
};

__device__ __forceinline__ v8f wmma_bf16(const Frag16& a, const Frag16& b, v8f c) {
    return __builtin_amdgcn_wmma_f32_16x16x32_bf16(
        false, a.v, false, b.v, (short)0, c, false, false);
}

// ---------------------------------------------------------------------------
// Kernel 0: zero output accumulator + per-expert counters
// ---------------------------------------------------------------------------
__global__ __launch_bounds__(256) void init_kernel(float* out, int* counts) {
    const size_t i = (size_t)blockIdx.x * 256 + threadIdx.x;   // grid covers B*D
    out[i] = 0.f;
    if (blockIdx.x == 0 && threadIdx.x < E_EXP) counts[threadIdx.x] = 0;
}

// ---------------------------------------------------------------------------
// Kernel 1: x fp32 -> bf16 (vectorized), so FFN A-frags load bf16 directly
// ---------------------------------------------------------------------------
__global__ __launch_bounds__(256) void cvt_x_kernel(
    const float* __restrict__ x, __bf16* __restrict__ xb)
{
    const size_t i = ((size_t)blockIdx.x * 256 + threadIdx.x) * 4;
    const float4 f = *(const float4*)(x + i);
    xb[i + 0] = (__bf16)f.x;
    xb[i + 1] = (__bf16)f.y;
    xb[i + 2] = (__bf16)f.z;
    xb[i + 3] = (__bf16)f.w;
}

// ---------------------------------------------------------------------------
// Kernel 2: tiled transpose + fp32->bf16 convert.
//   in : [E, R, C] fp32      out: [E, C, R] bf16   (K-major for wmma B frags)
// ---------------------------------------------------------------------------
__global__ __launch_bounds__(256) void transpose_cvt_kernel(
    const float* __restrict__ in, __bf16* __restrict__ out, int R, int C)
{
    const int tilesR = R >> 5, tilesC = C >> 5;
    const int perE   = tilesR * tilesC;
    const int e   = blockIdx.x / perE;
    const int rem = blockIdx.x % perE;
    const int rt  = rem / tilesC, ct = rem % tilesC;

    __shared__ float t[32][33];
    const int c  = threadIdx.x & 31;
    const int r0 = threadIdx.x >> 5;

#pragma unroll
    for (int i = 0; i < 4; ++i) {
        const int r = r0 + 8 * i;
        t[r][c] = in[((size_t)e * R + rt * 32 + r) * C + ct * 32 + c];
    }
    __syncthreads();
#pragma unroll
    for (int i = 0; i < 4; ++i) {
        const int r = r0 + 8 * i;
        out[((size_t)e * C + ct * 32 + r) * R + rt * 32 + c] = (__bf16)t[c][r];
    }
}

// ---------------------------------------------------------------------------
// Kernel 3: gating. One block (256 thr) per token.
// ---------------------------------------------------------------------------
__global__ __launch_bounds__(256) void gate_kernel(
    const float* __restrict__ x, const float* __restrict__ Wg,
    const float* __restrict__ bg,
    float* __restrict__ scores,      // [B, E]
    int*   __restrict__ counts,      // [E]
    int*   __restrict__ tok_list,    // [E, B]
    float* __restrict__ tok_scale)   // [E, B]
{
    const int b   = blockIdx.x;
    const int tid = threadIdx.x;
    __shared__ float red[E_EXP * 256];
    __shared__ float sc[E_EXP];
    __shared__ int   top2[TOP_K];
    __shared__ float tsc[TOP_K];

    float acc[E_EXP];
#pragma unroll
    for (int e = 0; e < E_EXP; ++e) acc[e] = 0.f;

    const float* xr = x + (size_t)b * D_DIM;
    for (int d = tid; d < D_DIM; d += 256) {
        const float xv = xr[d];
        const float* wr = Wg + (size_t)d * E_EXP;
#pragma unroll
        for (int e = 0; e < E_EXP; ++e) acc[e] += xv * wr[e];
    }
#pragma unroll
    for (int e = 0; e < E_EXP; ++e) red[e * 256 + tid] = acc[e];
    __syncthreads();

    if (tid < E_EXP) {
        float s = 0.f;
        for (int t = 0; t < 256; ++t) s += red[tid * 256 + t];
        sc[tid] = s + bg[tid];
    }
    __syncthreads();

    if (tid == 0) {
        float mx = sc[0];
#pragma unroll
        for (int e = 1; e < E_EXP; ++e) mx = fmaxf(mx, sc[e]);
        float p[E_EXP], sum = 0.f;
#pragma unroll
        for (int e = 0; e < E_EXP; ++e) { p[e] = __expf(sc[e] - mx); sum += p[e]; }
        const float inv = 1.f / sum;
#pragma unroll
        for (int e = 0; e < E_EXP; ++e) { p[e] *= inv; sc[e] = p[e]; }
        int i0 = 0;                       // ties -> lower index (lax.top_k)
#pragma unroll
        for (int e = 1; e < E_EXP; ++e) if (p[e] > p[i0]) i0 = e;
        int i1 = (i0 == 0) ? 1 : 0;
#pragma unroll
        for (int e = 0; e < E_EXP; ++e) if (e != i0 && p[e] > p[i1]) i1 = e;
        const float denom = p[i0] + p[i1];
        top2[0] = i0; top2[1] = i1;
        tsc[0] = p[i0] / denom; tsc[1] = p[i1] / denom;
    }
    __syncthreads();

    if (tid < E_EXP) scores[(size_t)b * E_EXP + tid] = sc[tid];
    if (tid < TOP_K) {
        const int e   = top2[tid];
        const int pos = atomicAdd(&counts[e], 1);
        tok_list [(size_t)e * B_TOK + pos] = b;
        tok_scale[(size_t)e * B_TOK + pos] = tsc[tid];
    }
}

// ---------------------------------------------------------------------------
// Kernel 4: grouped expert FFN, bf16 WMMA, 32-token tiles (2M x 2N outer
// product per k-step: 4 wmma from 4 fragment loads).
//   grid.x = E * (B/32). block = 256 threads = 8 wave32.
//   LDS: hid[32][4096] bf16 = 256 KB (CDNA5 320KB WGP LDS) + tile meta.
//   GEMM1 A-frags come straight from global bf16 x (tiny per-block working
//   set, WGP$-resident); B-frags are K-major bf16 -> all loads are b128.
//   Output: atomicAdd (exactly 2 adds/element onto zeroed out -> deterministic).
// ---------------------------------------------------------------------------
__global__ __launch_bounds__(256) void ffn_kernel(
    const __bf16* __restrict__ xb,    // [B, D] bf16
    const __bf16* __restrict__ W1t,   // [E, H, D] bf16
    const float*  __restrict__ b1w,
    const __bf16* __restrict__ W2t,   // [E, D, H] bf16
    const float*  __restrict__ b2w,
    const int*    __restrict__ counts,
    const int*    __restrict__ tok_list,
    const float*  __restrict__ tok_scale,
    float*        __restrict__ out)   // [B, D], pre-zeroed
{
    const int tilesPerExpert = B_TOK / TM;           // 256
    const int e    = blockIdx.x / tilesPerExpert;
    const int tile = blockIdx.x % tilesPerExpert;
    const int cnt  = counts[e];
    const int row0 = tile * TM;
    if (row0 >= cnt) return;
    int valid = cnt - row0; if (valid > TM) valid = TM;

    extern __shared__ char smem[];
    __bf16* hid    = (__bf16*)smem;                        // 32*4096*2 = 262144
    int*    stok   = (int*)  (smem + 262144);
    float*  sscale = (float*)(smem + 262144 + 128);

    const int tid  = threadIdx.x;
    const int wave = tid >> 5;
    const int lane = tid & 31;
    const int g    = lane >> 4;
    const int n    = lane & 15;
    const int g8   = g << 3;

    if (tid < TM) {
        int tok = 0; float scl = 0.f;
        const int idx = row0 + tid;
        if (idx < cnt) {
            tok = tok_list [(size_t)e * B_TOK + idx];
            scl = tok_scale[(size_t)e * B_TOK + idx];
        }
        stok[tid] = tok; sscale[tid] = scl;
    }
    __syncthreads();

    // per-lane gathered A-row pointers (token rows for M-tiles 0 and 1)
    const __bf16* arow0 = xb + (size_t)stok[n]      * D_DIM;
    const __bf16* arow1 = xb + (size_t)stok[16 + n] * D_DIM;

    // ---------------- GEMM1: hid[32,4096] = relu(x @ W1e + b1) --------------
#pragma unroll 1
    for (int nt = 0; nt < 32; nt += 2) {
        const int h0 = wave * 512 + nt * 16;
        v8f c00 = {0.f,0.f,0.f,0.f,0.f,0.f,0.f,0.f};
        v8f c01 = c00, c10 = c00, c11 = c00;
        const __bf16* brow0 = W1t + ((size_t)e * H_DIM + h0 + n) * D_DIM;
        const __bf16* brow1 = brow0 + (size_t)16 * D_DIM;
#pragma unroll 1
        for (int k0 = 0; k0 < D_DIM; k0 += 32) {
            Frag16 a0, a1, fb0, fb1;
            a0.q[0]  = *(const uint4*)(arow0 + k0 + g8);
            a0.q[1]  = *(const uint4*)(arow0 + k0 + 16 + g8);
            a1.q[0]  = *(const uint4*)(arow1 + k0 + g8);
            a1.q[1]  = *(const uint4*)(arow1 + k0 + 16 + g8);
            fb0.q[0] = *(const uint4*)(brow0 + k0 + g8);
            fb0.q[1] = *(const uint4*)(brow0 + k0 + 16 + g8);
            fb1.q[0] = *(const uint4*)(brow1 + k0 + g8);
            fb1.q[1] = *(const uint4*)(brow1 + k0 + 16 + g8);
            c00 = wmma_bf16(a0, fb0, c00);
            c10 = wmma_bf16(a1, fb0, c10);
            c01 = wmma_bf16(a0, fb1, c01);
            c11 = wmma_bf16(a1, fb1, c11);
        }
        const float biasA = b1w[(size_t)e * H_DIM + h0 + n];
        const float biasB = b1w[(size_t)e * H_DIM + h0 + 16 + n];
#pragma unroll
        for (int r = 0; r < 8; ++r) {
            const size_t rA = (size_t)(g8 + r) * H_DIM;
            const size_t rB = (size_t)(16 + g8 + r) * H_DIM;
            hid[rA + h0 + n]      = (__bf16)fmaxf(c00[r] + biasA, 0.f);
            hid[rB + h0 + n]      = (__bf16)fmaxf(c10[r] + biasA, 0.f);
            hid[rA + h0 + 16 + n] = (__bf16)fmaxf(c01[r] + biasB, 0.f);
            hid[rB + h0 + 16 + n] = (__bf16)fmaxf(c11[r] + biasB, 0.f);
        }
    }
    __syncthreads();

    // ---------------- GEMM2: out += scale * (hid @ W2e + b2) ----------------
    const __bf16* harow0 = hid + (size_t)n        * H_DIM;
    const __bf16* harow1 = hid + (size_t)(16 + n) * H_DIM;
#pragma unroll 1
    for (int dt = 0; dt < 8; dt += 2) {
        const int d0 = wave * 128 + dt * 16;
        v8f c00 = {0.f,0.f,0.f,0.f,0.f,0.f,0.f,0.f};
        v8f c01 = c00, c10 = c00, c11 = c00;
        const __bf16* brow0 = W2t + ((size_t)e * D_DIM + d0 + n) * H_DIM;
        const __bf16* brow1 = brow0 + (size_t)16 * H_DIM;
#pragma unroll 1
        for (int k0 = 0; k0 < H_DIM; k0 += 32) {
            Frag16 a0, a1, fb0, fb1;
            a0.q[0]  = *(const uint4*)(harow0 + k0 + g8);
            a0.q[1]  = *(const uint4*)(harow0 + k0 + 16 + g8);
            a1.q[0]  = *(const uint4*)(harow1 + k0 + g8);
            a1.q[1]  = *(const uint4*)(harow1 + k0 + 16 + g8);
            fb0.q[0] = *(const uint4*)(brow0 + k0 + g8);
            fb0.q[1] = *(const uint4*)(brow0 + k0 + 16 + g8);
            fb1.q[0] = *(const uint4*)(brow1 + k0 + g8);
            fb1.q[1] = *(const uint4*)(brow1 + k0 + 16 + g8);
            c00 = wmma_bf16(a0, fb0, c00);
            c10 = wmma_bf16(a1, fb0, c10);
            c01 = wmma_bf16(a0, fb1, c01);
            c11 = wmma_bf16(a1, fb1, c11);
        }
        const float biasA = b2w[(size_t)e * D_DIM + d0 + n];
        const float biasB = b2w[(size_t)e * D_DIM + d0 + 16 + n];
#pragma unroll
        for (int r = 0; r < 8; ++r) {
            const int m0 = g8 + r;
            const int m1 = 16 + g8 + r;
            if (m0 < valid) {
                float* dst = out + (size_t)stok[m0] * D_DIM + d0 + n;
                unsafeAtomicAdd(dst,      sscale[m0] * (c00[r] + biasA));
                unsafeAtomicAdd(dst + 16, sscale[m0] * (c01[r] + biasB));
            }
            if (m1 < valid) {
                float* dst = out + (size_t)stok[m1] * D_DIM + d0 + n;
                unsafeAtomicAdd(dst,      sscale[m1] * (c10[r] + biasA));
                unsafeAtomicAdd(dst + 16, sscale[m1] * (c11[r] + biasB));
            }
        }
    }
}

// ---------------------------------------------------------------------------
// Kernel 5: deterministic load-balance loss -> out[B*D]
// ---------------------------------------------------------------------------
__global__ __launch_bounds__(256) void loss_kernel(
    const float* __restrict__ scores, float* __restrict__ loss_slot)
{
    __shared__ float red[E_EXP * 256];
    const int tid = threadIdx.x;
    float acc[E_EXP];
#pragma unroll
    for (int e = 0; e < E_EXP; ++e) acc[e] = 0.f;
    for (int b = tid; b < B_TOK; b += 256) {
        const float* s = scores + (size_t)b * E_EXP;
#pragma unroll
        for (int e = 0; e < E_EXP; ++e) acc[e] += s[e];
    }
#pragma unroll
    for (int e = 0; e < E_EXP; ++e) red[e * 256 + tid] = acc[e];
    __syncthreads();
    if (tid < E_EXP) {
        float s = 0.f;
        for (int t = 0; t < 256; ++t) s += red[tid * 256 + t];
        red[tid * 256] = s;
    }
    __syncthreads();
    if (tid == 0) {
        float l = 0.f;
#pragma unroll
        for (int e = 0; e < E_EXP; ++e) {
            const float u = red[e * 256] / (float)B_TOK;
            l += u * u;
        }
        *loss_slot = PENALTY * l;
    }
}

// ---------------------------------------------------------------------------
extern "C" void kernel_launch(void* const* d_in, const int* in_sizes, int n_in,
                              void* d_out, int out_size, void* d_ws, size_t ws_size,
                              hipStream_t stream) {
    const float* x  = (const float*)d_in[0];
    const float* Wg = (const float*)d_in[1];
    const float* bg = (const float*)d_in[2];
    const float* W1 = (const float*)d_in[3];
    const float* b1 = (const float*)d_in[4];
    const float* W2 = (const float*)d_in[5];
    const float* b2 = (const float*)d_in[6];
    float* out = (float*)d_out;                       // [B*D] output + [1] loss

    // Workspace layout (needs ~145 MB):
    //   [0,32)        counts
    //   [4K, ...)     tok_list (256KB), tok_scale (256KB), scores (256KB)
    //   [1M, 17M)     x bf16           B*D*2   = 16 MB
    //   [17M, 81M)    W1t bf16         E*H*D*2 = 64 MB
    //   [81M, 145M)   W2t bf16         E*D*H*2 = 64 MB
    char* ws = (char*)d_ws;
    const size_t EB4  = (size_t)E_EXP * B_TOK * 4;
    const size_t WSZ  = (size_t)E_EXP * D_DIM * H_DIM * 2;
    int*    counts    = (int*)ws;
    int*    tok_list  = (int*)  (ws + 4096);
    float*  tok_scale = (float*)(ws + 4096 + EB4);
    float*  scores    = (float*)(ws + 4096 + 2 * EB4);
    __bf16* xb        = (__bf16*)(ws + (1u << 20));
    __bf16* W1t       = (__bf16*)(ws + (1u << 20) + (size_t)B_TOK * D_DIM * 2);
    __bf16* W2t       = (__bf16*)(ws + (1u << 20) + (size_t)B_TOK * D_DIM * 2 + WSZ);

    const int nOut = B_TOK * D_DIM;                    // 8,388,608 (div by 256)
    init_kernel<<<nOut / 256, 256, 0, stream>>>(out, counts);

    cvt_x_kernel<<<nOut / 1024, 256, 0, stream>>>(x, xb);

    // K-major bf16 weight copies: W1 [E,D,H] -> [E,H,D]; W2 [E,H,D] -> [E,D,H]
    transpose_cvt_kernel<<<E_EXP * (D_DIM / 32) * (H_DIM / 32), 256, 0, stream>>>(
        W1, W1t, D_DIM, H_DIM);
    transpose_cvt_kernel<<<E_EXP * (H_DIM / 32) * (D_DIM / 32), 256, 0, stream>>>(
        W2, W2t, H_DIM, D_DIM);

    gate_kernel<<<B_TOK, 256, 0, stream>>>(x, Wg, bg, scores, counts,
                                           tok_list, tok_scale);

    loss_kernel<<<1, 256, 0, stream>>>(scores, out + (size_t)B_TOK * D_DIM);

    const int ffnGrid = E_EXP * (B_TOK / TM);          // 2048 blocks
    const size_t ffnSmem = 262144 + 256;               // 256 KB hid + tile meta
    ffn_kernel<<<ffnGrid, 256, ffnSmem, stream>>>(xb, W1t, b1, W2t, b2,
                                                  counts, tok_list, tok_scale,
                                                  out);
}
